// KGAT_76501957476481
// MI455X (gfx1250) — matrix operand order repaired and checked by reading, not compiled, not run on hardware
//
#include <hip/hip_runtime.h>
#include <hip/hip_bf16.h>

typedef _Float16 v8h  __attribute__((ext_vector_type(8)));
typedef _Float16 v16h __attribute__((ext_vector_type(16)));
typedef float    v8f  __attribute__((ext_vector_type(8)));
typedef unsigned int u32x4 __attribute__((ext_vector_type(4)));
typedef int          i32x4 __attribute__((ext_vector_type(4)));
typedef int          i32x8 __attribute__((ext_vector_type(8)));

#define WAVES   2
#define DIM     64
#define KN      64
#define ALPHA   0.2f
#define NEGINF  (-9.0e15f)
#define EPSF    1e-8f

#if defined(__has_builtin)
#if __has_builtin(__builtin_amdgcn_tensor_load_to_lds) && __has_builtin(__builtin_amdgcn_s_wait_tensorcnt)
#define USE_TDM 1
#endif
#endif
#ifndef USE_TDM
#define USE_TDM 0
#endif

#if USE_TDM
// Issue one TDM load of a 64x64 f32 tile (16KB) from global into LDS.
// D# per CDNA5 ISA: group0 = {flags, lds_addr, gaddr_lo,
// gaddr_hi|type=2}; group1 = {data_size=4B, tensor_dim0/1=64, tile_dim0/1=64,
// tensor_dim0_stride=64}; groups 2/3 zero (2-D tensor). Tracked by TENSORcnt.
__device__ __forceinline__ void tdm_load_64x64_f32(void* lds_dst, const float* gsrc) {
  unsigned long long ga = (unsigned long long)(uintptr_t)gsrc;
  unsigned ga_lo = (unsigned)__builtin_amdgcn_readfirstlane((int)(unsigned)ga);
  unsigned ga_hi = (unsigned)__builtin_amdgcn_readfirstlane((int)(unsigned)(ga >> 32));
  unsigned lds   = (unsigned)__builtin_amdgcn_readfirstlane((int)(unsigned)(uintptr_t)lds_dst);

  u32x4 g0;
  g0[0] = 1u;                                   // count=1, user descriptor
  g0[1] = lds;                                  // lds_addr (low 32 bits = LDS offset)
  g0[2] = ga_lo;                                // global_addr[31:0]
  g0[3] = (ga_hi & 0x01FFFFFFu) | (2u << 30);   // global_addr[56:32] | type=2 ("image")

  i32x8 g1;
  g1[0] = 0x00020000;                           // workgroup_mask=0, data_size=2 (4 bytes)
  g1[1] = (DIM << 16);                          // tensor_dim0 = 64 (low 16 @ bits 63:48)
  g1[2] = (DIM << 16);                          // tensor_dim0_hi=0 | tensor_dim1 = 64
  g1[3] = (DIM << 16);                          // tensor_dim1_hi=0 | tile_dim0 = 64
  g1[4] = KN;                                   // tile_dim1 = 64, tile_dim2 = 0
  g1[5] = DIM;                                  // tensor_dim0_stride = 64 (low 32)
  g1[6] = 0;
  g1[7] = 0;

  i32x4 gz;
  gz[0] = 0; gz[1] = 0; gz[2] = 0; gz[3] = 0;
#if __clang_major__ >= 23
  i32x8 gz8;
  for (int i = 0; i < 8; ++i) gz8[i] = 0;
  __builtin_amdgcn_tensor_load_to_lds(g0, g1, gz, gz, gz8, 0);
#else
  __builtin_amdgcn_tensor_load_to_lds(g0, g1, gz, gz, 0);
#endif
}
#endif

__device__ __forceinline__ v16h make_frag(const _Float16* p0, const _Float16* p1) {
  v8h lo = *(const v8h*)p0;   // 16B -> ds_load_b128
  v8h hi = *(const v8h*)p1;
  return __builtin_shufflevector(lo, hi, 0,1,2,3,4,5,6,7,8,9,10,11,12,13,14,15);
}

__global__ __launch_bounds__(WAVES * 32) void kgat_layer(
    const float* __restrict__ src, const float* __restrict__ dst,
    const float* __restrict__ rel, const float* __restrict__ fcw,
    const float* __restrict__ fcb, const int* __restrict__ mask,
    float* __restrict__ out, int nnodes)
{
  __shared__ __align__(16) _Float16 s_dst[WAVES][KN * DIM]; // per-wave dst tile, f16 (8KB ea)
  __shared__ __align__(16) _Float16 s_w2[DIM * DIM];        // s_w2[d*64+f] = fcw[d][64+f]
#if USE_TDM
  __shared__ __align__(16) float    s_rel[WAVES][KN * DIM]; // per-wave rel tile, f32 (16KB ea)
#endif
  __shared__ float s_hsb[WAVES][DIM];                       // src@W1^T + bias
  __shared__ float s_att[WAVES][KN];

  const int tid  = threadIdx.x;
  const int lane = tid & 31;
  const int wave = tid >> 5;
  const int col  = lane & 15;
  const int hlf  = lane >> 4;

  const int node = blockIdx.x * WAVES + wave;     // wave-uniform
  const bool active = node < nnodes;              // wave-uniform -> EXEC all-ones inside
  const size_t nb = (size_t)node * (KN * DIM);

#if USE_TDM
  // Kick off the async rel-tile fetch first: overlaps with dst staging + setup.
  if (active) tdm_load_64x64_f32(&s_rel[wave][0], rel + nb);
#endif

  // Stage W2 (f16, transposed use) once per block; L2-resident across blocks.
  for (int i = tid; i < DIM * DIM; i += WAVES * 32) {
    int d = i >> 6, f = i & 63;
    s_w2[i] = (_Float16)fcw[d * 128 + 64 + f];
  }

  // Stage dst[n] (64x64 f32 -> f16 in LDS), coalesced b128 loads
  if (active) {
    const float4* p = (const float4*)(dst + nb);
    #pragma unroll
    for (int i = 0; i < 32; ++i) {
      float4 v = p[lane + i * 32];
      int idx = (lane + i * 32) * 4;
      s_dst[wave][idx + 0] = (_Float16)v.x;
      s_dst[wave][idx + 1] = (_Float16)v.y;
      s_dst[wave][idx + 2] = (_Float16)v.z;
      s_dst[wave][idx + 3] = (_Float16)v.w;
    }
  }
  __syncthreads();

  if (active) {
    // hsrc[d] = b[d] + sum_f src[n,f] * W1[d,f]  (k-independent, folded into WMMA C)
    {
      const float* srow = src + (size_t)node * DIM;
      int d0 = lane * 2;
      float a0 = fcb[d0], a1 = fcb[d0 + 1];
      for (int f = 0; f < DIM; ++f) {
        float sv = srow[f];
        a0 = fmaf(sv, fcw[d0 * 128 + f], a0);
        a1 = fmaf(sv, fcw[(d0 + 1) * 128 + f], a1);
      }
      s_hsb[wave][d0]     = a0;
      s_hsb[wave][d0 + 1] = a1;
    }

    // B fragments: B[f,d] = W2[d,f]; lane holds column (nt*16+col), K contiguous
    v16h bf[2][4];
    #pragma unroll
    for (int fb = 0; fb < 2; ++fb)
      #pragma unroll
      for (int nt = 0; nt < 4; ++nt) {
        const _Float16* p = &s_w2[(nt * 16 + col) * 64 + fb * 32 + hlf * 16];
        bf[fb][nt] = make_frag(p, p + 8);
      }

#if USE_TDM
    __builtin_amdgcn_s_wait_tensorcnt(0);   // rel tile now resident in LDS
#else
    const float* reln = rel + nb;
#endif

    #pragma unroll
    for (int kt = 0; kt < 4; ++kt) {
      // A fragments from LDS: row = neighbor, K runs contiguous (two b128 loads each)
      v16h af[2];
      const int row = kt * 16 + col;
      #pragma unroll
      for (int fb = 0; fb < 2; ++fb) {
        const _Float16* p = &s_dst[wave][row * 64 + fb * 32 + hlf * 16];
        af[fb] = make_frag(p, p + 8);
      }

      // C init = hsrc[d] (bias included): constant along tile rows
      v8f acc[4];
      #pragma unroll
      for (int nt = 0; nt < 4; ++nt) {
        float ci = s_hsb[wave][nt * 16 + col];
        #pragma unroll
        for (int i = 0; i < 8; ++i) acc[nt][i] = ci;
      }
      #pragma unroll
      for (int nt = 0; nt < 4; ++nt) {
        acc[nt] = __builtin_amdgcn_wmma_f32_16x16x32_f16(
            false, af[0], false, bf[0][nt], (short)0, acc[nt], false, false);
        acc[nt] = __builtin_amdgcn_wmma_f32_16x16x32_f16(
            false, af[1], false, bf[1][nt], (short)0, acc[nt], false, false);
      }

      // cosine(h, rel) per neighbor row; C layout: VGPR r, lane -> (M=r+8*hlf, N=nt*16+col)
      #pragma unroll
      for (int r = 0; r < 8; ++r) {
        int m = kt * 16 + r + hlf * 8;
        float dt = 0.f, h2 = 0.f, r2 = 0.f;
        #pragma unroll
        for (int nt = 0; nt < 4; ++nt) {
          float hv = acc[nt][r];
#if USE_TDM
          float rv = s_rel[wave][m * 64 + nt * 16 + col];
#else
          float rv = reln[m * 64 + nt * 16 + col];
#endif
          dt = fmaf(hv, rv, dt);
          h2 = fmaf(hv, hv, h2);
          r2 = fmaf(rv, rv, r2);
        }
        #pragma unroll
        for (int s = 1; s < 16; s <<= 1) {  // 16-lane butterfly (stays within half-group)
          dt += __shfl_xor(dt, s, 32);
          h2 += __shfl_xor(h2, s, 32);
          r2 += __shfl_xor(r2, s, 32);
        }
        if (col == 0) {
          float e = dt / (fmaxf(sqrtf(h2), EPSF) * fmaxf(sqrtf(r2), EPSF));
          e = (e > 0.f) ? e : ALPHA * e;                        // leaky_relu
          int mv = mask[(size_t)node * KN + m];
          s_att[wave][m] = (mv > 0) ? e : NEGINF;
        }
      }
    }

    // softmax over K=64 (2 values/lane, wave32 butterflies)
    {
      float e0 = s_att[wave][lane];
      float e1 = s_att[wave][lane + 32];
      float mx = fmaxf(e0, e1);
      #pragma unroll
      for (int s = 1; s < 32; s <<= 1) mx = fmaxf(mx, __shfl_xor(mx, s, 32));
      float p0 = __expf(e0 - mx), p1 = __expf(e1 - mx);
      float sm = p0 + p1;
      #pragma unroll
      for (int s = 1; s < 32; s <<= 1) sm += __shfl_xor(sm, s, 32);
      float inv = 1.0f / sm;
      s_att[wave][lane]      = p0 * inv;
      s_att[wave][lane + 32] = p1 * inv;
    }

    // agg[d] = sum_k att[k]*dst[n,k,d] + src[n,d]  (dst reused from LDS, no re-read)
    {
      int d0 = lane * 2;
      float a0 = 0.f, a1 = 0.f;
      #pragma unroll 8
      for (int k = 0; k < KN; ++k) {
        float w = s_att[wave][k];
        a0 = fmaf(w, (float)s_dst[wave][k * 64 + d0], a0);
        a1 = fmaf(w, (float)s_dst[wave][k * 64 + d0 + 1], a1);
      }
      float2 o;
      o.x = a0 + src[(size_t)node * DIM + d0];
      o.y = a1 + src[(size_t)node * DIM + d0 + 1];
      *(float2*)(out + (size_t)node * DIM + d0) = o;
    }
  }
}

extern "C" void kernel_launch(void* const* d_in, const int* in_sizes, int n_in,
                              void* d_out, int out_size, void* d_ws, size_t ws_size,
                              hipStream_t stream) {
  const float* src = (const float*)d_in[0];
  const float* dst = (const float*)d_in[1];
  const float* rel = (const float*)d_in[2];
  const float* fcw = (const float*)d_in[3];
  const float* fcb = (const float*)d_in[4];
  const int*   msk = (const int*)d_in[5];
  float* out = (float*)d_out;

  const int n = in_sizes[0] / DIM;              // N = 50000
  const int blocks = (n + WAVES - 1) / WAVES;

  // Layer 1: src -> out.  Layer 2: out -> out (each node reads only its own row).
  kgat_layer<<<blocks, WAVES * 32, 0, stream>>>(src, dst, rel, fcw, fcb, msk, out, n);
  kgat_layer<<<blocks, WAVES * 32, 0, stream>>>(out, dst, rel, fcw, fcb, msk, out, n);
}
